// LCAOInteraction_16037407883306
// MI455X (gfx1250) — matrix operand order repaired and checked by reading, not compiled, not run on hardware
//
#include <hip/hip_runtime.h>
#include <hip/hip_bf16.h>

// ---------------------------------------------------------------------------
// LCAO interaction layer for MI455X (gfx1250, wave32).
// Memory-bound problem: c (E*9*128) dominates. Stored bf16. Updated-c is
// algebraically eliminated: einsum(rbw, c*(1+twt)) = (1+twt) * einsum(rbw, c).
// All dense layers use v_wmma_f32_16x16x32_bf16 (f32 accumulate).
// Fast-math activations via v_rcp_f32 / v_rsq_f32 (no IEEE div sequences).
// ---------------------------------------------------------------------------

#define NN_   20000
#define EE_   80000
#define TT_   120000
#define ORB_  9
#define HID_  128
#define CONV_ 128

typedef __attribute__((ext_vector_type(16))) __bf16       v16bf;
typedef __attribute__((ext_vector_type(8)))  float        v8f;
typedef __attribute__((ext_vector_type(8)))  unsigned int v8u;

// float -> bf16 bits, round-to-nearest-even
__device__ __forceinline__ unsigned short f2bf_u(float f) {
  unsigned u; __builtin_memcpy(&u, &f, 4);
  unsigned r = u + 0x7FFFu + ((u >> 16) & 1u);
  return (unsigned short)(r >> 16);
}
__device__ __forceinline__ float bf2f(unsigned short h) {
  unsigned u = ((unsigned)h) << 16;
  float f; __builtin_memcpy(&f, &u, 4);
  return f;
}
__device__ __forceinline__ float fast_rcp(float x) {
  return __builtin_amdgcn_rcpf(x);           // v_rcp_f32
}
__device__ __forceinline__ float fast_rsq(float x) {
  return __builtin_amdgcn_rsqf(x);           // v_rsq_f32
}
__device__ __forceinline__ float sigmoid_f(float x) {
  return fast_rcp(1.f + __expf(-x));
}
__device__ __forceinline__ float silu_f(float x) {
  return x * sigmoid_f(x);
}

// ---------------------------------------------------------------------------
// One 16x16 output tile of  D += A(16xK) * W(nbase..nbase+15, 0..K-1)^T
// A: bf16 (ushort bits), row-major in LDS, row stride = astride halves.
// W: bf16 weights (n_out x K row-major) in global memory (L2 resident).
// A 16x32 k order used here (contiguous, not the ISA table order): lane l (m = l&15, hi = l>>4):
//   vgpr v holds K pair at  k = 16*hi + 2*v
// B 32x16 k order used here (same contiguous order as A): lane l (n = l&15, hi = l>>4):
//   vgpr v holds K pair at  k = 16*hi + 2*v
// C/D f32 16x16: lane l, vgpr r -> (m = r + 8*hi, n = l&15)
// ---------------------------------------------------------------------------
__device__ __forceinline__ void gemm_tile(const unsigned short* A, int astride,
                                          const unsigned short* W, int K,
                                          int nbase, v8f& acc) {
  const int l  = threadIdx.x & 31;
  const int mn = l & 15;
  const int hi = l >> 4;
  const unsigned short* wrow = W + (size_t)(nbase + mn) * K;
  for (int k0 = 0; k0 < K; k0 += 32) {
    v8u au, bu;
#pragma unroll
    for (int v = 0; v < 8; ++v) {
      const int ka = k0 + (hi << 4) + (v << 1);
      au[v] = *(const unsigned int*)(A + mn * astride + ka);
      const int kb = k0 + (hi << 4) + (v << 1);
      bu[v] = *(const unsigned int*)(wrow + kb);
    }
    acc = __builtin_amdgcn_wmma_f32_16x16x32_bf16(
        false, __builtin_bit_cast(v16bf, au),
        false, __builtin_bit_cast(v16bf, bu),
        (short)0, acc, false, false);
  }
}

// Cooperative tile loader: nfloat4 float4s from global f32 -> packed bf16 LDS.
// Emits global_load_b128 + ds_store_b64 (two packed u32 per float4).
__device__ __forceinline__ void load_tile_bf16(const float* __restrict__ src,
                                               unsigned short* dst,
                                               int nfloat4) {
  const float4* s4 = (const float4*)src;
  unsigned* d = (unsigned*)dst;
  for (int i = threadIdx.x; i < nfloat4; i += 256) {
    const float4 v = s4[i];
    d[i * 2]     = (unsigned)f2bf_u(v.x) | ((unsigned)f2bf_u(v.y) << 16);
    d[i * 2 + 1] = (unsigned)f2bf_u(v.z) | ((unsigned)f2bf_u(v.w) << 16);
  }
}

// ---------------------------------------------------------------------------
// prep: f32 -> bf16 weight conversion
// ---------------------------------------------------------------------------
__global__ void k_f2bf(const float* __restrict__ src,
                       unsigned short* __restrict__ dst, int n) {
  int i = blockIdx.x * 256 + threadIdx.x;
  if (i < n) dst[i] = f2bf_u(src[i]);
}

__global__ void k_rbw(const float* __restrict__ rb,
                      const float* __restrict__ cw,
                      float* __restrict__ rbw) {
  int i = blockIdx.x * 256 + threadIdx.x;
  if (i < EE_ * ORB_) rbw[i] = rb[i] * cw[i / ORB_];
}

// ---------------------------------------------------------------------------
// h = x @ Wn^T + bn ; split -> x_e (f32), sigmoid(xk) (f32)
// grid: N/16 blocks of 256 threads (8 waves, each wave does 2 N-tiles).
// ---------------------------------------------------------------------------
__global__ __launch_bounds__(256) void k_node_linear(
    const float* __restrict__ x, const unsigned short* __restrict__ Wnb,
    const float* __restrict__ bn, float* __restrict__ x_e,
    float* __restrict__ xk_sig) {
  __shared__ unsigned short Abf[16 * 128];
  const int mbase = blockIdx.x * 16;
  load_tile_bf16(x + (size_t)mbase * 128, Abf, 512);
  __syncthreads();
  const int w = threadIdx.x >> 5, l = threadIdx.x & 31;
  const int nn = l & 15, hi = l >> 4;
#pragma unroll
  for (int t = 0; t < 2; ++t) {
    const int nbase = (w * 2 + t) * 16;
    v8f acc = {};
    gemm_tile(Abf, 128, Wnb, 128, nbase, acc);
    const float bias = bn[nbase + nn];
#pragma unroll
    for (int r = 0; r < 8; ++r) {
      const float v = acc[r] + bias;
      const size_t row = mbase + r + 8 * hi;
      if (nbase < 128)
        x_e[row * 128 + nbase + nn] = v;
      else
        xk_sig[row * 128 + (nbase - 128) + nn] = sigmoid_f(v);
    }
  }
}

// ---------------------------------------------------------------------------
// c = silu(silu(cji @ Wc1^T) @ Wc2^T), rows = E*9, output bf16
// grid: (E*9)/16 blocks, 256 threads, wave w owns N-tile w.
// ---------------------------------------------------------------------------
__global__ __launch_bounds__(256) void k_coeff_mlp(
    const float* __restrict__ cji, const unsigned short* __restrict__ Wc1b,
    const unsigned short* __restrict__ Wc2b,
    unsigned short* __restrict__ c_bf) {
  __shared__ unsigned short Abf[16 * 128];
  __shared__ unsigned short Hbf[16 * 128];
  const size_t mbase = (size_t)blockIdx.x * 16;
  load_tile_bf16(cji + mbase * 128, Abf, 512);
  __syncthreads();
  const int w = threadIdx.x >> 5, l = threadIdx.x & 31;
  const int nn = l & 15, hi = l >> 4;
  const int nbase = w * 16;
  v8f acc = {};
  gemm_tile(Abf, 128, Wc1b, 128, nbase, acc);
#pragma unroll
  for (int r = 0; r < 8; ++r)
    Hbf[(r + 8 * hi) * 128 + nbase + nn] = f2bf_u(silu_f(acc[r]));
  __syncthreads();
  v8f acc2 = {};
  gemm_tile(Hbf, 128, Wc2b, 128, nbase, acc2);
#pragma unroll
  for (int r = 0; r < 8; ++r)
    c_bf[(mbase + r + 8 * hi) * 128 + nbase + nn] = f2bf_u(silu_f(acc2[r]));
}

// ---------------------------------------------------------------------------
// s[e,:] = sum_d rbw[e,d] * c[e,d,:]        (streaming, contiguous)
// grid: E/2 blocks of 256 (2 edges / block, thread = channel)
// ---------------------------------------------------------------------------
__global__ __launch_bounds__(256) void k_lcao_pre(
    const unsigned short* __restrict__ c_bf, const float* __restrict__ rbw,
    float* __restrict__ s) {
  const int e = blockIdx.x * 2 + (threadIdx.x >> 7);
  const int h = threadIdx.x & 127;
  const unsigned short* cp = c_bf + (size_t)e * ORB_ * 128 + h;
  const float* wp = rbw + (size_t)e * ORB_;
  float acc = 0.f;
#pragma unroll
  for (int d = 0; d < ORB_; ++d) acc += wp[d] * bf2f(cp[d * 128]);
  s[(size_t)e * 128 + h] = acc;
}

// ---------------------------------------------------------------------------
// Triplet stage: tw = l2norm(sum_d (rbw[kj,d]*shb[t,d]) * c[kj,d,:])
//                     * sigmoid(xk[k]) ; segment_sum over edge_idx_ji.
// grid: T/2 blocks of 256 (2 triplets / block, 128 threads each = 4 waves)
// ---------------------------------------------------------------------------
__global__ __launch_bounds__(256) void k_triplet(
    const unsigned short* __restrict__ c_bf, const float* __restrict__ rbw,
    const float* __restrict__ shb, const float* __restrict__ xk_sig,
    const int* __restrict__ tri_idx_k, const int* __restrict__ edge_idx_kj,
    const int* __restrict__ edge_idx_ji, float* __restrict__ tw_edge) {
  __shared__ float ssum[2][4];
  const int g = threadIdx.x >> 7;
  const int h = threadIdx.x & 127;
  const int t = blockIdx.x * 2 + g;
  const int kj = edge_idx_kj[t];
  const int ji = edge_idx_ji[t];
  const int kn = tri_idx_k[t];
  const unsigned short* cp = c_bf + (size_t)kj * ORB_ * 128 + h;
  const float* xkp = xk_sig + (size_t)kn * 128 + h;
  // issue random-address cachelines early (global_prefetch_b8); the c[kj]
  // gather is the latency-dominant random read over a >L2 working set
  __builtin_prefetch(cp, 0, 0);
  __builtin_prefetch(cp + 4 * 128, 0, 0);
  __builtin_prefetch(xkp, 0, 0);
  const float* rp = rbw + (size_t)kj * ORB_;
  const float* sp = shb + (size_t)t * ORB_;
  float acc = 0.f;
#pragma unroll
  for (int d = 0; d < ORB_; ++d) acc += rp[d] * sp[d] * bf2f(cp[d * 128]);
  float sq = acc * acc;
#pragma unroll
  for (int m = 16; m >= 1; m >>= 1) sq += __shfl_xor(sq, m, 32);
  if ((threadIdx.x & 31) == 0) ssum[g][(threadIdx.x >> 5) & 3] = sq;
  __syncthreads();
  const float tot = ssum[g][0] + ssum[g][1] + ssum[g][2] + ssum[g][3];
  const float inv = fast_rsq(fmaxf(tot, 1e-24f));
  const float val = acc * inv * (*xkp);
  atomicAdd(&tw_edge[(size_t)ji * 128 + h], val);
}

// ---------------------------------------------------------------------------
// Edge stage (fused): twt = tw @ Wt^T ; u = s*(1+twt) ; lcao = l2norm(u)
// fn = silu(silu([x_e[i],x_e[j]] @ Wf1^T + bf1) @ Wf2^T + bf2)
// msg = (lcao @ Wb^T) * fn ; atomic segment_sum into node_acc[idx_i]
// grid: E/16 blocks, 256 threads, wave w owns N-tile w.
// ---------------------------------------------------------------------------
__global__ __launch_bounds__(256) void k_edge(
    const float* __restrict__ tw_edge, const float* __restrict__ s,
    const float* __restrict__ x_e, const int* __restrict__ idx_i,
    const int* __restrict__ idx_j, const unsigned short* __restrict__ Wtb,
    const unsigned short* __restrict__ Wbb,
    const unsigned short* __restrict__ Wf1b, const float* __restrict__ bf1,
    const unsigned short* __restrict__ Wf2b, const float* __restrict__ bf2,
    float* __restrict__ node_acc) {
  __shared__ unsigned short TWbf[16 * 128];   // tw tile, later reused as fn-hidden
  __shared__ unsigned short XEbf[16 * 256];   // concat(x_e[i], x_e[j])
  __shared__ unsigned short LCbf[16 * 128];   // normalized lcao
  __shared__ float partial[16][8];
  __shared__ float invn[16];
  __shared__ int rows_i[16];
  const int mbase = blockIdx.x * 16;

  load_tile_bf16(tw_edge + (size_t)mbase * 128, TWbf, 512);
  {
    unsigned* d = (unsigned*)XEbf;
    for (int i = threadIdx.x; i < 1024; i += 256) {   // 16 rows x 64 float4
      const int m = i >> 6, q = i & 63;
      const int src = (q < 32) ? idx_i[mbase + m] : idx_j[mbase + m];
      const float4 v =
          *(const float4*)(x_e + (size_t)src * 128 + ((q & 31) << 2));
      d[i * 2]     = (unsigned)f2bf_u(v.x) | ((unsigned)f2bf_u(v.y) << 16);
      d[i * 2 + 1] = (unsigned)f2bf_u(v.z) | ((unsigned)f2bf_u(v.w) << 16);
    }
  }
  if (threadIdx.x < 16) rows_i[threadIdx.x] = idx_i[mbase + threadIdx.x];
  __syncthreads();

  const int w = threadIdx.x >> 5, l = threadIdx.x & 31;
  const int nn = l & 15, hi = l >> 4;
  const int nbase = w * 16;

  // --- twt + u = s*(1+twt), per-row sum of squares ---
  v8f twt = {};
  gemm_tile(TWbf, 128, Wtb, 128, nbase, twt);
  v8f u;
#pragma unroll
  for (int r = 0; r < 8; ++r) {
    const size_t row = mbase + r + 8 * hi;
    const float uv = s[row * 128 + nbase + nn] * (1.f + twt[r]);
    u[r] = uv;
    float pr = uv * uv;                     // reduce over 16 n-lanes of half
    pr += __shfl_xor(pr, 1, 32);
    pr += __shfl_xor(pr, 2, 32);
    pr += __shfl_xor(pr, 4, 32);
    pr += __shfl_xor(pr, 8, 32);
    if (nn == 0) partial[r + 8 * hi][w] = pr;
  }
  __syncthreads();
  if (threadIdx.x < 16) {
    float tot = 0.f;
#pragma unroll
    for (int j = 0; j < 8; ++j) tot += partial[threadIdx.x][j];
    invn[threadIdx.x] = fast_rsq(fmaxf(tot, 1e-24f));
  }
  __syncthreads();
#pragma unroll
  for (int r = 0; r < 8; ++r)
    LCbf[(r + 8 * hi) * 128 + nbase + nn] = f2bf_u(u[r] * invn[r + 8 * hi]);
  __syncthreads();

  // --- fn layer 1 (K = 256), hidden into TWbf (safe: TW reads are done) ---
  {
    v8f a;
    const float b = bf1[nbase + nn];
#pragma unroll
    for (int r = 0; r < 8; ++r) a[r] = b;
    gemm_tile(XEbf, 256, Wf1b, 256, nbase, a);
#pragma unroll
    for (int r = 0; r < 8; ++r)
      TWbf[(r + 8 * hi) * 128 + nbase + nn] = f2bf_u(silu_f(a[r]));
  }
  __syncthreads();

  // --- fn layer 2 and lcao @ Wb^T ---
  v8f fn;
  {
    const float b = bf2[nbase + nn];
#pragma unroll
    for (int r = 0; r < 8; ++r) fn[r] = b;
    gemm_tile(TWbf, 128, Wf2b, 128, nbase, fn);
#pragma unroll
    for (int r = 0; r < 8; ++r) fn[r] = silu_f(fn[r]);
  }
  v8f g = {};
  gemm_tile(LCbf, 128, Wbb, 128, nbase, g);

#pragma unroll
  for (int r = 0; r < 8; ++r) {
    const float msg = g[r] * fn[r];
    const int dst = rows_i[r + 8 * hi];
    atomicAdd(&node_acc[(size_t)dst * 128 + nbase + nn], msg);
  }
}

// ---------------------------------------------------------------------------
// out = x + node_acc @ Wo^T      grid: N/16 blocks, 256 threads
// ---------------------------------------------------------------------------
__global__ __launch_bounds__(256) void k_out(
    const float* __restrict__ node_acc, const float* __restrict__ x,
    const unsigned short* __restrict__ Wob, float* __restrict__ out) {
  __shared__ unsigned short Abf[16 * 128];
  const int mbase = blockIdx.x * 16;
  load_tile_bf16(node_acc + (size_t)mbase * 128, Abf, 512);
  __syncthreads();
  const int w = threadIdx.x >> 5, l = threadIdx.x & 31;
  const int nn = l & 15, hi = l >> 4;
  const int nbase = w * 16;
  v8f acc = {};
  gemm_tile(Abf, 128, Wob, 128, nbase, acc);
#pragma unroll
  for (int r = 0; r < 8; ++r) {
    const size_t row = mbase + r + 8 * hi;
    out[row * 128 + nbase + nn] = x[row * 128 + nbase + nn] + acc[r];
  }
}

// ---------------------------------------------------------------------------
extern "C" void kernel_launch(void* const* d_in, const int* in_sizes, int n_in,
                              void* d_out, int out_size, void* d_ws,
                              size_t ws_size, hipStream_t stream) {
  const float* x        = (const float*)d_in[0];
  const float* cji      = (const float*)d_in[1];
  const float* cutoff_w = (const float*)d_in[2];
  const float* rb       = (const float*)d_in[3];
  const float* shb      = (const float*)d_in[4];
  const int*   idx_i    = (const int*)d_in[5];
  const int*   idx_j    = (const int*)d_in[6];
  const int*   tri_k    = (const int*)d_in[7];
  const int*   e_kj     = (const int*)d_in[8];
  const int*   e_ji     = (const int*)d_in[9];
  const float* Wn  = (const float*)d_in[10];
  const float* bn  = (const float*)d_in[11];
  const float* Wc1 = (const float*)d_in[12];
  const float* Wc2 = (const float*)d_in[13];
  const float* Wt  = (const float*)d_in[14];
  const float* Wb  = (const float*)d_in[15];
  const float* Wf1 = (const float*)d_in[16];
  const float* bf1 = (const float*)d_in[17];
  const float* Wf2 = (const float*)d_in[18];
  const float* bf2 = (const float*)d_in[19];
  const float* Wo  = (const float*)d_in[20];
  float* out = (float*)d_out;

  // workspace layout (bytes), ~300 MB total
  char* w = (char*)d_ws;
  size_t off = 0;
  auto alloc = [&](size_t bytes) {
    void* p = w + off;
    off = (off + bytes + 255) & ~(size_t)255;
    return p;
  };
  float* x_e     = (float*)alloc((size_t)NN_ * 128 * 4);
  float* xk_sig  = (float*)alloc((size_t)NN_ * 128 * 4);
  float* s_buf   = (float*)alloc((size_t)EE_ * 128 * 4);
  float* tw_edge = (float*)alloc((size_t)EE_ * 128 * 4);
  float* nacc    = (float*)alloc((size_t)NN_ * 128 * 4);
  float* rbw     = (float*)alloc((size_t)EE_ * ORB_ * 4);
  unsigned short* c_bf = (unsigned short*)alloc((size_t)EE_ * ORB_ * 128 * 2);
  unsigned short* Wnb  = (unsigned short*)alloc(256 * 128 * 2);
  unsigned short* Wc1b = (unsigned short*)alloc(128 * 128 * 2);
  unsigned short* Wc2b = (unsigned short*)alloc(128 * 128 * 2);
  unsigned short* Wtb  = (unsigned short*)alloc(128 * 128 * 2);
  unsigned short* Wbb  = (unsigned short*)alloc(128 * 128 * 2);
  unsigned short* Wf1b = (unsigned short*)alloc(128 * 256 * 2);
  unsigned short* Wf2b = (unsigned short*)alloc(128 * 128 * 2);
  unsigned short* Wob  = (unsigned short*)alloc(128 * 128 * 2);

  // prep: bf16 weights, rbw, zeroed accumulators
  auto cvt = [&](const float* src, unsigned short* dst, int n) {
    k_f2bf<<<(n + 255) / 256, 256, 0, stream>>>(src, dst, n);
  };
  cvt(Wn, Wnb, 256 * 128);
  cvt(Wc1, Wc1b, 128 * 128);
  cvt(Wc2, Wc2b, 128 * 128);
  cvt(Wt, Wtb, 128 * 128);
  cvt(Wb, Wbb, 128 * 128);
  cvt(Wf1, Wf1b, 128 * 256);
  cvt(Wf2, Wf2b, 128 * 128);
  cvt(Wo, Wob, 128 * 128);
  k_rbw<<<(EE_ * ORB_ + 255) / 256, 256, 0, stream>>>(rb, cutoff_w, rbw);
  hipMemsetAsync(tw_edge, 0, (size_t)EE_ * 128 * 4, stream);
  hipMemsetAsync(nacc, 0, (size_t)NN_ * 128 * 4, stream);

  // pipeline
  k_node_linear<<<NN_ / 16, 256, 0, stream>>>(x, Wnb, bn, x_e, xk_sig);
  k_coeff_mlp<<<(EE_ * ORB_) / 16, 256, 0, stream>>>(cji, Wc1b, Wc2b, c_bf);
  k_lcao_pre<<<EE_ / 2, 256, 0, stream>>>(c_bf, rbw, s_buf);
  k_triplet<<<TT_ / 2, 256, 0, stream>>>(c_bf, rbw, shb, xk_sig, tri_k, e_kj,
                                         e_ji, tw_edge);
  k_edge<<<EE_ / 16, 256, 0, stream>>>(tw_edge, s_buf, x_e, idx_i, idx_j, Wtb,
                                       Wbb, Wf1b, bf1, Wf2b, bf2, nacc);
  k_out<<<NN_ / 16, 256, 0, stream>>>(nacc, x, Wob, out);
}
